// GNN_Bet8_18485539242355
// MI455X (gfx1250) — matrix operand order repaired and checked by reading, not compiled, not run on hardware
//
#include <hip/hip_runtime.h>

// ---------------------------------------------------------------------------
// CDNA5 (gfx1250) GNN_Bet kernel. Dense GEMMs use V_WMMA_F32_16X16X32_BF16
// (bf16 A/B, f32 accumulate). Tiles are staged in LDS pre-permuted into the
// ISA 7.12.2 fragment layout so each lane loads its fragment with two
// ds_load_b128 ops. GEMM is templated on NOUT so the WMMA tile loop is fully
// static (no runtime guards -> no accumulator phi-copy storms). SpMM uses f32
// global atomics (accumulator lives in the 192MB L2). Wave32 throughout.
// ---------------------------------------------------------------------------

typedef __attribute__((ext_vector_type(16))) __bf16 v16bf;
typedef __attribute__((ext_vector_type(8)))  float  v8f;

union Frag16 { v16bf v; unsigned short u[16]; uint4 q[2]; };

__device__ __forceinline__ unsigned short f2bf(float f) {
  unsigned int u = __float_as_uint(f);
  u += 0x7FFFu + ((u >> 16) & 1u);   // round-to-nearest-even
  return (unsigned short)(u >> 16);
}
__device__ __forceinline__ float bf2f(unsigned short h) {
  return __uint_as_float(((unsigned int)h) << 16);
}

// ---------------------------------------------------------------------------
// GEMM: out = act(A_bf16[M,K] @ W_f32[K,NOUT] + bias)
// Block = 256 threads (8 waves). Block tile = 32 rows x NOUT cols.
// Wave w: row-group rg = w>>2 (16 rows), col-wave cw = w&3; wave owns
// NT = NOUT/64 accumulator tiles (tile j = cw + 4t), so one A-fragment feeds
// NT WMMAs per k-step. K stepped by 32.
//
// LDS fragment layout (inverting ISA 7.12.2):
//   A (16x32 bf16):  lane = m + 16*((k>>3)&1),  half h = (k&7) | ((k&16)>>1)
//   B (32x16 bf16):  lane = n + 16*((k>>3)&1),  half h = (k&7) | ((k&16)>>1)
// Rows padded to 24 halfs (48B): 16B-aligned, 12-bank stagger -> conflict-free
// ds_load_b128 fragment fetches.
// ---------------------------------------------------------------------------
template <int NOUT>
__global__ __launch_bounds__(256) void gemm_bf16_kernel(
    const unsigned short* __restrict__ A,   // [M,K] bf16 row-major
    const float* __restrict__ W,            // [K,NOUT] f32 row-major
    const float* __restrict__ bias,         // [NOUT] or null
    float* __restrict__ outF,               // [M,NOUT] f32 or null
    unsigned short* __restrict__ outB,      // [M,NOUT] bf16 or null
    int M, int K, int relu)
{
  constexpr int NT  = NOUT / 64;            // accumulator tiles per wave
  constexpr int LNN = (NOUT == 256) ? 8 : 7;

  __shared__ __align__(16) unsigned short Afrag[2][32][24];
  __shared__ __align__(16) unsigned short Bfrag[NOUT / 16][32][24];

  const int tid  = threadIdx.x;
  const int wv   = tid >> 5;
  const int lane = tid & 31;
  const int rg   = wv >> 2;      // row-group 0..1
  const int cw   = wv & 3;       // col-wave 0..3
  const int g    = lane >> 4;    // half-wave select
  const int m    = lane & 15;
  const int row0 = blockIdx.x * 32;

  v8f acc[NT];
#pragma unroll
  for (int t = 0; t < NT; ++t)
    acc[t] = (v8f){0.f, 0.f, 0.f, 0.f, 0.f, 0.f, 0.f, 0.f};

  for (int k0 = 0; k0 < K; k0 += 32) {
    // ---- stage A tile: 32 rows x 32 K bf16 (512 dwords), permuted ----
#pragma unroll
    for (int i0 = 0; i0 < 2; ++i0) {
      int i = tid + i0 * 256;
      int r = i >> 4;                       // 0..31
      int k = (i & 15) * 2;                 // even K
      int grow = row0 + r; grow = grow < M ? grow : M - 1;  // clamp, no branch
      unsigned int val = *(const unsigned int*)(A + (size_t)grow * K + k0 + k);
      int ga = (k >> 3) & 1;
      int h  = (k & 7) | ((k & 16) >> 1);
      ((unsigned int*)&Afrag[r >> 4][(r & 15) + 16 * ga][0])[h >> 1] = val;
    }
    // ---- stage W tile: 32 K x NOUT, f32->bf16, permuted (pairs of K) ----
#pragma unroll
    for (int i0 = 0; i0 < (NOUT << 4) / 256; ++i0) {
      int i  = tid + i0 * 256;
      int c  = i & (NOUT - 1);
      int kr = ((i >> LNN) << 1);           // even K row 0..30
      float w0 = W[(size_t)(k0 + kr) * NOUT + c];
      float w1 = W[(size_t)(k0 + kr + 1) * NOUT + c];
      unsigned int pk = (unsigned int)f2bf(w0) | ((unsigned int)f2bf(w1) << 16);
      int lb = (c & 15) + 16 * ((kr >> 3) & 1);
      ((unsigned int*)&Bfrag[c >> 4][lb][0])[((kr & 7) | ((kr & 16) >> 1)) >> 1] = pk;
    }
    __syncthreads();

    Frag16 a;
    a.q[0] = *(const uint4*)&Afrag[rg][lane][0];
    a.q[1] = *(const uint4*)&Afrag[rg][lane][8];
#pragma unroll
    for (int t = 0; t < NT; ++t) {
      Frag16 b;
      const int tj = cw + (t << 2);
      b.q[0] = *(const uint4*)&Bfrag[tj][lane][0];
      b.q[1] = *(const uint4*)&Bfrag[tj][lane][8];
      acc[t] = __builtin_amdgcn_wmma_f32_16x16x32_bf16(
          false, a.v, false, b.v, (short)0, acc[t], false, false);
    }
    __syncthreads();
  }

  // ---- epilogue: C/D layout -> VGPR r holds M = r + 8*g, N = m ----
#pragma unroll
  for (int t = 0; t < NT; ++t) {
    const int col = (cw + (t << 2)) * 16 + m;
    const float bv = bias ? bias[col] : 0.f;
#pragma unroll
    for (int r = 0; r < 8; ++r) {
      int row = row0 + rg * 16 + r + 8 * g;
      if (row < M) {
        float vv = acc[t][r] + bv;
        if (relu) vv = fmaxf(vv, 0.f);
        size_t idx = (size_t)row * NOUT + col;
        if (outF) outF[idx] = vv;
        if (outB) outB[idx] = f2bf(vv);
      }
    }
  }
}

// ---------------------------------------------------------------------------
// SpMM: Y[r,:] += v * X[c,:]  (one wave per edge, 4 floats per lane, H=128)
// ---------------------------------------------------------------------------
__global__ __launch_bounds__(256) void spmm_kernel(
    const int* __restrict__ rows, const int* __restrict__ cols,
    const float* __restrict__ vals, const float* __restrict__ X,
    float* __restrict__ Y, int E)
{
  int idx  = blockIdx.x * 256 + threadIdx.x;
  int e    = idx >> 5;
  int lane = idx & 31;
  if (e >= E) return;
  int r = rows[e], c = cols[e];
  float v = vals[e];
  float4 d = ((const float4*)(X + (size_t)c * 128))[lane];
  float* yp = Y + (size_t)r * 128 + lane * 4;
  atomicAdd(yp + 0, v * d.x);
  atomicAdd(yp + 1, v * d.y);
  atomicAdd(yp + 2, v * d.z);
  atomicAdd(yp + 3, v * d.w);
}

// ---------------------------------------------------------------------------
// relu (+ optional row L2-norm), emit bf16 copy. One wave per row (H=128).
// ---------------------------------------------------------------------------
__global__ __launch_bounds__(256) void relu_norm_kernel(
    const float* __restrict__ Y, unsigned short* __restrict__ Xb,
    int Nrows, int do_norm)
{
  int idx  = blockIdx.x * 256 + threadIdx.x;
  int row  = idx >> 5;
  int lane = idx & 31;
  if (row >= Nrows) return;
  float4 d = ((const float4*)(Y + (size_t)row * 128))[lane];
  d.x = fmaxf(d.x, 0.f); d.y = fmaxf(d.y, 0.f);
  d.z = fmaxf(d.z, 0.f); d.w = fmaxf(d.w, 0.f);
  float s = d.x * d.x + d.y * d.y + d.z * d.z + d.w * d.w;
  for (int off = 16; off > 0; off >>= 1) s += __shfl_xor(s, off, 32);
  float scale = 1.f;
  if (do_norm) scale = 1.f / fmaxf(sqrtf(s), 1e-12f);
  unsigned int lo = (unsigned int)f2bf(d.x * scale) |
                    ((unsigned int)f2bf(d.y * scale) << 16);
  unsigned int hi = (unsigned int)f2bf(d.z * scale) |
                    ((unsigned int)f2bf(d.w * scale) << 16);
  uint2 pk; pk.x = lo; pk.y = hi;
  ((uint2*)(Xb + (size_t)row * 128))[lane] = pk;
}

// ---------------------------------------------------------------------------
// score op: score[row] (+)= H2[row,:] . m3 + b3   (256-dot, one wave per row)
// ---------------------------------------------------------------------------
__global__ __launch_bounds__(256) void dot_score_kernel(
    const unsigned short* __restrict__ H2, const float* __restrict__ m3,
    const float* __restrict__ b3, float* __restrict__ score,
    int Nrows, int accum)
{
  int idx  = blockIdx.x * 256 + threadIdx.x;
  int row  = idx >> 5;
  int lane = idx & 31;
  if (row >= Nrows) return;
  const unsigned short* h = H2 + (size_t)row * 256;
  float s = 0.f;
#pragma unroll
  for (int j = 0; j < 8; ++j) {
    int c = lane + j * 32;
    s += bf2f(h[c]) * m3[c];
  }
  for (int off = 16; off > 0; off >>= 1) s += __shfl_xor(s, off, 32);
  if (lane == 0) {
    float r = s + b3[0];
    score[row] = accum ? (score[row] + r) : r;
  }
}

__global__ void mul_kernel(const float* __restrict__ a,
                           const float* __restrict__ b,
                           float* __restrict__ out, int n) {
  int i = blockIdx.x * blockDim.x + threadIdx.x;
  if (i < n) out[i] = a[i] * b[i];
}

// ---------------------------------------------------------------------------
extern "C" void kernel_launch(void* const* d_in, const int* in_sizes, int n_in,
                              void* d_out, int out_size, void* d_ws, size_t ws_size,
                              hipStream_t stream) {
  (void)n_in; (void)out_size; (void)ws_size;
  const int H = 128;
  const int* a1r = (const int*)d_in[0];
  const int* a1c = (const int*)d_in[1];
  const float* a1v = (const float*)d_in[2];
  const int* a2r = (const int*)d_in[3];
  const int* a2c = (const int*)d_in[4];
  const float* a2v = (const float*)d_in[5];
  const int E1 = in_sizes[0], E2 = in_sizes[3];
  const float* w1 = (const float*)d_in[6];
  const int N = in_sizes[6] / H;
  const float* wlist[8];
  for (int i = 0; i < 8; ++i) wlist[i] = (const float*)d_in[7 + i];
  const float* m1 = (const float*)d_in[15];
  const float* b1 = (const float*)d_in[16];
  const float* m2 = (const float*)d_in[17];
  const float* b2 = (const float*)d_in[18];
  const float* m3 = (const float*)d_in[19];
  const float* b3 = (const float*)d_in[20];

  // Workspace carve (aliasing: t<->h1, y<->h2 have disjoint live ranges)
  char* p = (char*)d_ws;
  auto carve = [&](size_t bytes) {
    char* q = p; p += (bytes + 255) & ~(size_t)255; return q;
  };
  unsigned short* xb = (unsigned short*)carve((size_t)N * H * 2); // bf16 features
  float*          t  = (float*)carve((size_t)N * H * 4);          // x@W (f32)
  unsigned short* h1 = (unsigned short*)t;                        // [N,256] bf16
  float*          y  = (float*)carve((size_t)N * H * 4);          // spmm accum
  unsigned short* h2 = (unsigned short*)y;                        // [N,256] bf16
  float* s1 = (float*)carve((size_t)N * 4);
  float* s2 = (float*)carve((size_t)N * 4);

  const int rowBlocks = (N + 7) / 8;       // wave-per-row kernels
  const int gemmGrid  = (N + 31) / 32;     // 32 rows per GEMM block

  auto mlp = [&](float* score, int accum) {
    gemm_bf16_kernel<256><<<gemmGrid, 256, 0, stream>>>(
        xb, m1, b1, nullptr, h1, N, 128, 1);
    gemm_bf16_kernel<256><<<gemmGrid, 256, 0, stream>>>(
        h1, m2, b2, nullptr, h2, N, 256, 1);
    dot_score_kernel<<<rowBlocks, 256, 0, stream>>>(h2, m3, b3, score, N, accum);
  };

  auto branch = [&](const int* rows, const int* cols, const float* vals,
                    int E, float* score) {
    hipMemsetAsync(y, 0, (size_t)N * H * 4, stream);
    spmm_kernel<<<(E + 7) / 8, 256, 0, stream>>>(rows, cols, vals, w1, y, E);
    relu_norm_kernel<<<rowBlocks, 256, 0, stream>>>(y, xb, N, 1);
    mlp(score, 0);
    for (int i = 0; i < 8; ++i) {
      gemm_bf16_kernel<128><<<gemmGrid, 256, 0, stream>>>(
          xb, wlist[i], nullptr, t, nullptr, N, 128, 0);
      hipMemsetAsync(y, 0, (size_t)N * H * 4, stream);
      spmm_kernel<<<(E + 7) / 8, 256, 0, stream>>>(rows, cols, vals, t, y, E);
      relu_norm_kernel<<<rowBlocks, 256, 0, stream>>>(y, xb, N, (i < 7) ? 1 : 0);
      mlp(score, 1);
    }
  };

  branch(a1r, a1c, a1v, E1, s1);
  branch(a2r, a2c, a2v, E2, s2);
  mul_kernel<<<(N + 255) / 256, 256, 0, stream>>>(s1, s2, (float*)d_out, N);
}
